// Mamba2Block_21251498180799
// MI455X (gfx1250) — compile-verified, matrix operand order checked by reading, not run on hardware
//
#include <hip/hip_runtime.h>
#include <hip/hip_bf16.h>
#include <math.h>

typedef __bf16 bf16;
typedef __attribute__((ext_vector_type(16))) __bf16 v16bf;
typedef __attribute__((ext_vector_type(8)))  float  v8f;

#define B_   4
#define S_   2048
#define D_   1024
#define N_   64
#define K_   4
#define DI_  2048
#define H_   32
#define P_   64
#define ROW_ 4256          // 2*DI + H + 2*N
#define M_   (B_*S_)       // 8192
#define NPAD1_ 4352        // ROW_ padded to multiple of 128

// ---------------- weight transpose + fp32->bf16 convert (pads extra rows with 0)
__global__ void wt_kernel(const float* __restrict__ W, bf16* __restrict__ Wt,
                          int Kdim, int Ndim, int Npad) {
  long idx = (long)blockIdx.x * blockDim.x + threadIdx.x;
  long total = (long)Npad * Kdim;
  if (idx >= total) return;
  int n = (int)(idx / Kdim), k = (int)(idx % Kdim);
  float v = (n < Ndim) ? W[(size_t)k * Ndim + n] : 0.f;
  Wt[(size_t)n * Kdim + k] = (bf16)v;
}

// ---------------- bf16 WMMA GEMM: C[M,Nreal] = A[M,K](fp32) * Bt[Npad,K]^T
#define BM 128
#define BN 128
#define BK 32
#define LDS_S 40   // LDS row stride (bf16), 80B => 16B aligned rows, bank-skewed

__global__ __launch_bounds__(256)
void gemm_wmma(const float* __restrict__ A, int lda,
               const bf16* __restrict__ Bt,
               float* __restrict__ C, int ldc,
               int Kdim, int Nreal) {
  __shared__ bf16 As[BM * LDS_S];
  __shared__ bf16 Bs[BN * LDS_S];
  const int tid  = threadIdx.x;
  const int lane = tid & 31;
  const int w    = tid >> 5;    // 0..7 waves
  const int wm   = w >> 1;      // 0..3 -> M offset 32*wm
  const int wn   = w & 1;       // 0..1 -> N offset 64*wn
  const int blockM = blockIdx.y * BM;
  const int blockN = blockIdx.x * BN;

  v8f acc[2][4];
  v8f vzero = {};
  #pragma unroll
  for (int mi = 0; mi < 2; ++mi)
    #pragma unroll
    for (int ni = 0; ni < 4; ++ni) acc[mi][ni] = vzero;

  const int lrow = tid >> 1;         // 0..127
  const int lcol = (tid & 1) * 16;   // 0 or 16

  const int ktiles = Kdim / BK;
  for (int kt = 0; kt < ktiles; ++kt) {
    const int k0 = kt * BK;
    // stage A (convert fp32->bf16) and B (bf16 copy) into LDS
    {
      const float* src = A + (size_t)(blockM + lrow) * lda + k0 + lcol;
      bf16* dst = &As[lrow * LDS_S + lcol];
      #pragma unroll
      for (int j = 0; j < 16; ++j) dst[j] = (bf16)src[j];
      const bf16* bsrc = Bt + (size_t)(blockN + lrow) * Kdim + k0 + lcol;
      bf16* bdst = &Bs[lrow * LDS_S + lcol];
      #pragma unroll
      for (int j = 0; j < 16; ++j) bdst[j] = bsrc[j];
    }
    if (kt + 1 < ktiles) {   // gfx1250: lowers to global_prefetch_b8
      __builtin_prefetch(A  + (size_t)(blockM + lrow) * lda  + k0 + BK + lcol, 0, 1);
      __builtin_prefetch(Bt + (size_t)(blockN + lrow) * Kdim + k0 + BK + lcol, 0, 1);
    }
    __syncthreads();

    // ISA 7.12.2 bf16 A/B layout: lanes 0-15 -> K{0..7,16..23}, lanes 16-31 -> K{8..15,24..31}
    {
      const int r  = lane & 15;
      const int kb = (lane >> 4) << 3;  // 0 or 8
      v16bf afrag[2], bfrag[4];
      #pragma unroll
      for (int mi = 0; mi < 2; ++mi) {
        const bf16* ap = &As[(wm * 32 + mi * 16 + r) * LDS_S + kb];
        #pragma unroll
        for (int j = 0; j < 8; ++j) { afrag[mi][j] = ap[j]; afrag[mi][8 + j] = ap[16 + j]; }
      }
      #pragma unroll
      for (int ni = 0; ni < 4; ++ni) {
        const bf16* bp = &Bs[(wn * 64 + ni * 16 + r) * LDS_S + kb];
        #pragma unroll
        for (int j = 0; j < 8; ++j) { bfrag[ni][j] = bp[j]; bfrag[ni][8 + j] = bp[16 + j]; }
      }
      #pragma unroll
      for (int mi = 0; mi < 2; ++mi)
        #pragma unroll
        for (int ni = 0; ni < 4; ++ni)
          acc[mi][ni] = __builtin_amdgcn_wmma_f32_16x16x32_bf16(
              false, afrag[mi], false, bfrag[ni], (short)0, acc[mi][ni], false, false);
    }
    __syncthreads();
  }

  // C/D layout: vgpr rr -> M = rr + 8*(lane>=16); N = lane&15
  const int r = lane & 15;
  const int mhalf = (lane >> 4) * 8;
  #pragma unroll
  for (int mi = 0; mi < 2; ++mi)
    #pragma unroll
    for (int ni = 0; ni < 4; ++ni) {
      const int col = blockN + wn * 64 + ni * 16 + r;
      if (col < Nreal) {
        #pragma unroll
        for (int rr = 0; rr < 8; ++rr) {
          const int row = blockM + wm * 32 + mi * 16 + mhalf + rr;
          C[(size_t)row * ldc + col] = acc[mi][ni][rr];
        }
      }
    }
}

// ---------------- softplus(dt)+dA
__global__ void dt_kernel(const float* __restrict__ zx, const float* __restrict__ dt_bias,
                          const float* __restrict__ A_log,
                          float* __restrict__ dtsp, float* __restrict__ dAe) {
  int idx = blockIdx.x * blockDim.x + threadIdx.x;
  if (idx >= M_ * H_) return;
  int m = idx / H_, hh = idx % H_;
  float raw = zx[(size_t)m * ROW_ + 2 * DI_ + hh] + dt_bias[hh];
  float sp  = (raw > 20.f) ? raw : log1pf(expf(raw));
  dtsp[idx] = sp;
  dAe[idx]  = expf(-sp * expf(A_log[hh]));
}

// ---------------- depthwise causal conv1d + SiLU
__global__ void conv_kernel(const float* __restrict__ zx, const float* __restrict__ conv_w,
                            float* __restrict__ xconv) {
  long idx = (long)blockIdx.x * blockDim.x + threadIdx.x;
  if (idx >= (long)M_ * DI_) return;
  int m = (int)(idx / DI_), c = (int)(idx % DI_);
  int s = m % S_;
  float acc = 0.f;
  #pragma unroll
  for (int k = 0; k < K_; ++k) {
    int sk = s + k - (K_ - 1);
    if (sk >= 0)
      acc += zx[(size_t)(m + k - (K_ - 1)) * ROW_ + DI_ + c] * conv_w[c * K_ + k];
  }
  xconv[idx] = acc / (1.f + expf(-acc));   // silu
}

// ---------------- selective scan: state in VGPRs, B_t/C_t in LDS, wave32 shuffles
__global__ __launch_bounds__(128)
void scan_kernel(const float* __restrict__ zx, const float* __restrict__ xconv,
                 const float* __restrict__ dtsp, const float* __restrict__ dAe,
                 float* __restrict__ yg) {
  const int bid  = blockIdx.x;            // b*64 + h*2 + half
  const int half = bid & 1;
  const int h    = (bid >> 1) & (H_ - 1);
  const int b    = bid >> 6;
  const int t    = threadIdx.x;           // 0..127 (4 waves)
  const int p    = half * 32 + (t >> 2);
  const int q    = t & 3;
  const int nb   = q * 16;
  __shared__ float sB[N_], sC[N_];
  float hs[16];
  #pragma unroll
  for (int i = 0; i < 16; ++i) hs[i] = 0.f;

  for (int s = 0; s < S_; ++s) {
    const size_t m = (size_t)b * S_ + s;
    const float* row = zx + m * ROW_;
    if (t < 64)       sB[t]      = row[2 * DI_ + H_ + t];
    else if (t < 128) sC[t - 64] = row[2 * DI_ + H_ + N_ + (t - 64)];
    __syncthreads();
    const float da  = dAe[m * H_ + h];
    const float dt  = dtsp[m * H_ + h];
    const float xv  = xconv[m * DI_ + h * P_ + p];
    const float dtx = dt * xv;
    float y = 0.f;
    #pragma unroll
    for (int i = 0; i < 16; ++i) {
      hs[i] = da * hs[i] + dtx * sB[nb + i];
      y += hs[i] * sC[nb + i];
    }
    y += __shfl_xor(y, 1, 32);
    y += __shfl_xor(y, 2, 32);
    if (q == 0) {
      const float zv = row[h * P_ + p];
      yg[m * DI_ + h * P_ + p] = y * (zv / (1.f + expf(-zv)));  // gate with silu(z)
    }
    __syncthreads();
  }
}

// ---------------- residual + RMSNorm
__global__ __launch_bounds__(256)
void norm_kernel(const float* __restrict__ x, const float* __restrict__ proj,
                 const float* __restrict__ norm_w, float* __restrict__ out) {
  const int m = blockIdx.x;
  const int t = threadIdx.x;
  float r[4];
  float ss = 0.f;
  #pragma unroll
  for (int j = 0; j < 4; ++j) {
    const int i = t + j * 256;
    const float v = x[(size_t)m * D_ + i] + proj[(size_t)m * D_ + i];
    r[j] = v; ss += v * v;
  }
  #pragma unroll
  for (int off = 16; off >= 1; off >>= 1) ss += __shfl_xor(ss, off, 32);
  __shared__ float red[8];
  if ((t & 31) == 0) red[t >> 5] = ss;
  __syncthreads();
  if (t == 0) {
    float tot = 0.f;
    #pragma unroll
    for (int i = 0; i < 8; ++i) tot += red[i];
    red[0] = rsqrtf(tot / (float)D_ + 1e-6f);
  }
  __syncthreads();
  const float inv = red[0];
  #pragma unroll
  for (int j = 0; j < 4; ++j) {
    const int i = t + j * 256;
    out[(size_t)m * D_ + i] = r[j] * inv * norm_w[i];
  }
}

extern "C" void kernel_launch(void* const* d_in, const int* in_sizes, int n_in,
                              void* d_out, int out_size, void* d_ws, size_t ws_size,
                              hipStream_t stream) {
  const float* x      = (const float*)d_in[0];
  const float* W_in   = (const float*)d_in[1];
  const float* conv_w = (const float*)d_in[2];
  const float* dt_b   = (const float*)d_in[3];
  const float* A_log  = (const float*)d_in[4];
  const float* W_out  = (const float*)d_in[5];
  const float* norm_w = (const float*)d_in[6];
  float* out = (float*)d_out;

  char* ws = (char*)d_ws;
  auto alloc = [&](size_t bytes) -> char* {
    char* p = ws; ws += (bytes + 255) & ~(size_t)255; return p;
  };
  float* zx    = (float*)alloc((size_t)M_ * ROW_ * 4);
  float* xconv = (float*)alloc((size_t)M_ * DI_ * 4);
  float* yg    = (float*)alloc((size_t)M_ * DI_ * 4);
  float* proj  = (float*)alloc((size_t)M_ * D_  * 4);
  float* dtsp  = (float*)alloc((size_t)M_ * H_  * 4);
  float* dAe   = (float*)alloc((size_t)M_ * H_  * 4);
  bf16*  WtIn  = (bf16*) alloc((size_t)NPAD1_ * D_  * 2);
  bf16*  WtOut = (bf16*) alloc((size_t)D_     * DI_ * 2);

  // 1) weight transpose + bf16 convert (pad in_proj N to 4352)
  {
    long tot1 = (long)NPAD1_ * D_;
    wt_kernel<<<(tot1 + 255) / 256, 256, 0, stream>>>(W_in, WtIn, D_, ROW_, NPAD1_);
    long tot2 = (long)D_ * DI_;
    wt_kernel<<<(tot2 + 255) / 256, 256, 0, stream>>>(W_out, WtOut, DI_, D_, D_);
  }
  // 2) in_proj GEMM: zx = x @ W_in   (M=8192, K=1024, N=4256)
  {
    dim3 grid(NPAD1_ / BN, M_ / BM);
    gemm_wmma<<<grid, 256, 0, stream>>>(x, D_, WtIn, zx, ROW_, D_, ROW_);
  }
  // 3) dt: softplus + dA
  dt_kernel<<<(M_ * H_ + 255) / 256, 256, 0, stream>>>(zx, dt_b, A_log, dtsp, dAe);
  // 4) depthwise conv + silu
  {
    long tot = (long)M_ * DI_;
    conv_kernel<<<(int)((tot + 255) / 256), 256, 0, stream>>>(zx, conv_w, xconv);
  }
  // 5) selective scan + gating
  scan_kernel<<<B_ * H_ * 2, 128, 0, stream>>>(zx, xconv, dtsp, dAe, yg);
  // 6) out_proj GEMM: proj = yg @ W_out  (M=8192, K=2048, N=1024)
  {
    dim3 grid(D_ / BN, M_ / BM);
    gemm_wmma<<<grid, 256, 0, stream>>>(yg, DI_, WtOut, proj, D_, DI_, D_);
  }
  // 7) residual + RMSNorm
  norm_kernel<<<M_, 256, 0, stream>>>(x, proj, norm_w, out);
}
